// QGLU_39092792328542
// MI455X (gfx1250) — matrix operand order repaired and checked by reading, not compiled, not run on hardware
//
#include <hip/hip_runtime.h>
#include <hip/hip_bf16.h>

#define EMBED 2048
#define FFN   8192

typedef __attribute__((ext_vector_type(16))) __bf16 v16bf;
typedef __attribute__((ext_vector_type(8)))  __bf16 v8bf;
typedef __attribute__((ext_vector_type(8)))  float  v8f;
typedef __attribute__((ext_vector_type(4)))  float  v4f;

// Convert 4x float4 into a 16-wide bf16 fragment register set.
__device__ __forceinline__ v16bf cvt16(v4f a, v4f b, v4f c, v4f d) {
  v16bf r;
#pragma unroll
  for (int i = 0; i < 4; ++i) {
    r[i]      = (__bf16)a[i];
    r[4 + i]  = (__bf16)b[i];
    r[8 + i]  = (__bf16)c[i];
    r[12 + i] = (__bf16)d[i];
  }
  return r;
}

// B fragment: two 8-element K chunks 16 apart for this lane's column.
__device__ __forceinline__ v16bf loadB16(const float* __restrict__ p) {
  v4f a = *(const v4f*)(p);
  v4f b = *(const v4f*)(p + 4);
  v4f c = *(const v4f*)(p + 16);
  v4f d = *(const v4f*)(p + 20);
  return cvt16(a, b, c, d);
}

// A fragment: ISA bf16 A layout = two 8-element K chunks 16 apart.
__device__ __forceinline__ v16bf loadA16(const float* __restrict__ p) {
  v4f a = *(const v4f*)(p);
  v4f b = *(const v4f*)(p + 4);
  v4f c = *(const v4f*)(p + 16);
  v4f d = *(const v4f*)(p + 20);
  return cvt16(a, b, c, d);
}

__device__ __forceinline__ float gelu_exact(float v) {
  return 0.5f * v * (1.0f + erff(v * 0.70710678118654752f));
}

// Kernel 1: h[t,f] = gelu(x@W_fc1^T) * (x@W_gate^T), h stored bf16.
// Block = 256 threads = 8 waves; block tile 128M x 64N; wave tile 16M x 64N.
__global__ __launch_bounds__(256)
void glu_gemm1(const float* __restrict__ x, const float* __restrict__ Wg,
               const float* __restrict__ Wf, __bf16* __restrict__ h) {
  const int lane  = threadIdx.x & 31;
  const int wave  = threadIdx.x >> 5;
  const int Mbase = blockIdx.y * 128 + wave * 16;
  const int Nbase = blockIdx.x * 64;

  const int rowA = Mbase + (lane & 15);
  const int kShA = (lane >> 4) * 8;    // A: lanes 16-31 take the +8 K chunk
  const int kShB = (lane >> 4) * 8;    // B: lanes 16-31 take the +8 K chunk

  v8f accG[4], accF[4];
#pragma unroll
  for (int j = 0; j < 4; ++j) { accG[j] = (v8f)0.0f; accF[j] = (v8f)0.0f; }

  const float* __restrict__ xrow = x + (size_t)rowA * EMBED;

  for (int k0 = 0; k0 < EMBED; k0 += 32) {
    v16bf a = loadA16(xrow + k0 + kShA);
#pragma unroll
    for (int j = 0; j < 4; ++j) {
      const int col = Nbase + j * 16 + (lane & 15);
      const float* __restrict__ wgp = Wg + (size_t)col * EMBED + k0 + kShB;
      const float* __restrict__ wfp = Wf + (size_t)col * EMBED + k0 + kShB;
      // hint next K tile into cache (speculative prefetch, OOB dropped)
      __builtin_prefetch(wgp + 32, 0, 0);
      __builtin_prefetch(wfp + 32, 0, 0);
      v16bf bg = loadB16(wgp);
      v16bf bf = loadB16(wfp);
      accG[j] = __builtin_amdgcn_wmma_f32_16x16x32_bf16(
          false, a, false, bg, (short)0, accG[j], false, false);
      accF[j] = __builtin_amdgcn_wmma_f32_16x16x32_bf16(
          false, a, false, bf, (short)0, accF[j], false, false);
    }
  }

  // C/D layout: VGPR v, lanes 0-15 -> M=v, lanes 16-31 -> M=8+v; N = lane&15.
  const int mB = Mbase + (lane >> 4) * 8;
  const int nB = Nbase + (lane & 15);
#pragma unroll
  for (int j = 0; j < 4; ++j) {
#pragma unroll
    for (int v = 0; v < 8; ++v) {
      float val = gelu_exact(accF[j][v]) * accG[j][v];
      h[(size_t)(mB + v) * FFN + (nB + j * 16)] = (__bf16)val;
    }
  }
}

// Kernel 2: out[t,d] = h @ W_fc2^T ; h is bf16 [T,F], W_fc2 fp32 [D,F].
__global__ __launch_bounds__(256)
void glu_gemm2(const __bf16* __restrict__ h, const float* __restrict__ W2,
               float* __restrict__ out) {
  const int lane  = threadIdx.x & 31;
  const int wave  = threadIdx.x >> 5;
  const int Mbase = blockIdx.y * 128 + wave * 16;
  const int Nbase = blockIdx.x * 64;

  const int rowA = Mbase + (lane & 15);
  const int kShA = (lane >> 4) * 8;
  const int kShB = (lane >> 4) * 8;

  v8f acc[4];
#pragma unroll
  for (int j = 0; j < 4; ++j) acc[j] = (v8f)0.0f;

  const __bf16* __restrict__ hrow = h + (size_t)rowA * FFN;

  for (int k0 = 0; k0 < FFN; k0 += 32) {
    v8bf lo = *(const v8bf*)(hrow + k0 + kShA);
    v8bf hi = *(const v8bf*)(hrow + k0 + kShA + 16);
    v16bf a;
#pragma unroll
    for (int i = 0; i < 8; ++i) { a[i] = lo[i]; a[8 + i] = hi[i]; }
#pragma unroll
    for (int j = 0; j < 4; ++j) {
      const int col = Nbase + j * 16 + (lane & 15);
      const float* __restrict__ wp = W2 + (size_t)col * FFN + k0 + kShB;
      __builtin_prefetch(wp + 32, 0, 0);
      v16bf b = loadB16(wp);
      acc[j] = __builtin_amdgcn_wmma_f32_16x16x32_bf16(
          false, a, false, b, (short)0, acc[j], false, false);
    }
  }

  const int mB = Mbase + (lane >> 4) * 8;
  const int nB = Nbase + (lane & 15);
#pragma unroll
  for (int j = 0; j < 4; ++j) {
#pragma unroll
    for (int v = 0; v < 8; ++v) {
      out[(size_t)(mB + v) * EMBED + (nB + j * 16)] = acc[j][v];
    }
  }
}

extern "C" void kernel_launch(void* const* d_in, const int* in_sizes, int n_in,
                              void* d_out, int out_size, void* d_ws, size_t ws_size,
                              hipStream_t stream) {
  const float* x  = (const float*)d_in[0];   // [T, 2048]
  const float* Wg = (const float*)d_in[1];   // [8192, 2048]
  const float* Wf = (const float*)d_in[2];   // [8192, 2048]
  const float* W2 = (const float*)d_in[3];   // [2048, 8192]
  float* out = (float*)d_out;                // [T, 2048]
  __bf16* h = (__bf16*)d_ws;                 // [T, 8192] bf16 intermediate

  const int T = in_sizes[0] / EMBED;         // 8192

  dim3 blk(256);
  dim3 g1(FFN / 64, T / 128);
  glu_gemm1<<<g1, blk, 0, stream>>>(x, Wg, Wf, h);

  dim3 g2(EMBED / 64, T / 128);
  glu_gemm2<<<g2, blk, 0, stream>>>(h, W2, out);
}
